// GINConvNet_79611513798693
// MI455X (gfx1250) — matrix operand order repaired and checked by reading, not compiled, not run on hardware
//
#include <hip/hip_runtime.h>
#include <hip/hip_bf16.h>

// ---------------------------------------------------------------------------
// GIN network on gfx1250:
//   - f16 activations/weights, f32 WMMA accumulation (v_wmma_f32_16x16x32_f16)
//   - double-buffered async global->LDS tile staging (ASYNCcnt pipelining)
//   - fp32 atomic scatter for the GIN neighbor sum, fp32 BatchNorm statistics
// ---------------------------------------------------------------------------

typedef __attribute__((ext_vector_type(8)))  _Float16 v8h;
typedef __attribute__((ext_vector_type(16))) _Float16 v16h;
typedef __attribute__((ext_vector_type(8)))  float    v8f;

__device__ __forceinline__ v16h cat8(v8h lo, v8h hi) {
    return __builtin_shufflevector(lo, hi, 0,1,2,3,4,5,6,7,8,9,10,11,12,13,14,15);
}

__device__ __forceinline__ void atomAddF32(float* p, float v) {
    __hip_atomic_fetch_add(p, v, __ATOMIC_RELAXED, __HIP_MEMORY_SCOPE_AGENT);
}

// 16-byte async copy global -> LDS (per-lane, tracked by ASYNCcnt).
// Low 32 bits of a generic pointer to __shared__ are the LDS byte address.
__device__ __forceinline__ void async_load_b128(const void* gptr, void* lptr) {
    unsigned int lds = (unsigned int)(unsigned long long)lptr;
    asm volatile("global_load_async_to_lds_b128 %0, %1, off"
                 :: "v"(lds), "v"(gptr) : "memory");
}

// ------------------------------ zero fill ----------------------------------
__global__ void zero_k(float* __restrict__ p, int n) {
    int i = blockIdx.x * blockDim.x + threadIdx.x;
    if (i < n) p[i] = 0.0f;
}

// ----------------------- fp32 -> f16 convert -------------------------------
__global__ void cvt_k(const float* __restrict__ in, _Float16* __restrict__ out, int n) {
    int i = blockIdx.x * blockDim.x + threadIdx.x;
    if (i < n) out[i] = (_Float16)in[i];
}

// --------------------- h2 = h (slice copy, to fp32) ------------------------
template <typename T>
__global__ void copy_k(const T* __restrict__ X, int ldx, int xoff,
                       float* __restrict__ O, int ldo, int M, int d) {
    int idx = blockIdx.x * blockDim.x + threadIdx.x;
    if (idx >= M * d) return;
    int r = idx / d, c = idx % d;
    O[(size_t)r * ldo + c] = (float)X[(size_t)r * ldx + xoff + c];
}

// ------------------- h2[dst] += h[src]  (edge scatter) ---------------------
template <typename T>
__global__ void scatter_k(const T* __restrict__ H, int ldh, int hoff,
                          const int* __restrict__ src, const int* __restrict__ dst,
                          float* __restrict__ A, int lda, int E, int d4) {
    int idx = blockIdx.x * blockDim.x + threadIdx.x;
    if (idx >= E * d4) return;
    int e  = idx / d4;
    int c4 = (idx % d4) * 4;
    int s  = src[e];
    int t  = dst[e];
    const T* hp = H + (size_t)s * ldh + hoff + c4;
    float*   p  = A + (size_t)t * lda + c4;
    atomAddF32(p + 0, (float)hp[0]);
    atomAddF32(p + 1, (float)hp[1]);
    atomAddF32(p + 2, (float)hp[2]);
    atomAddF32(p + 3, (float)hp[3]);
}

// --------------------------- WMMA GEMM -------------------------------------
// Y[M x NC] = X[M x K] @ W[K x NC] + bias  (X, W in f16; f32 accumulation)
// Block: 256 threads = 8 waves. Block tile 128x64; wave tile 16x64 (4 WMMAs).
// Double-buffered async global->LDS staging. NC % 64 == 0, K % 32 == 0.
#define TILE_M 128
#define TILE_N 64
#define LDA_H  40   // 32 K-halfs + 8 pad  (80B rows, 16B-aligned chunks)
#define LDB_H  72   // 64 N-halfs + 8 pad  (144B rows, 16B-aligned chunks)

// Issue one tile's async copies: per thread 2 A-chunks + 1 B-chunk (16B each).
__device__ __forceinline__ void stage_tiles(
        const _Float16* __restrict__ X, int ldx,
        const _Float16* __restrict__ W, int NC, int n0,
        _Float16* lAbuf, _Float16* lBbuf,
        int t, int blockM, int M, int k0)
{
    #pragma unroll
    for (int i = 0; i < 2; ++i) {
        int id = t + i * 256;
        int r  = id >> 2;
        int c8 = (id & 3) * 8;
        int gr = blockM + r;
        if (gr >= M) gr = M - 1;   // clamp: rows beyond M are never stored
        async_load_b128(X + (size_t)gr * ldx + k0 + c8, &lAbuf[r * LDA_H + c8]);
    }
    {
        int k  = t >> 3;
        int c8 = (t & 7) * 8;
        async_load_b128(W + (size_t)(k0 + k) * NC + n0 + c8,
                        &lBbuf[k * LDB_H + c8]);
    }
}

__global__ __launch_bounds__(256) void wmma_gemm16(
        const _Float16* __restrict__ X, int ldx,
        const _Float16* __restrict__ W,      // K x NC row-major, f16
        const float* __restrict__ bias,
        float* __restrict__ Y32, _Float16* __restrict__ Y16,
        int ldy, int yoff,
        int M, int K, int NC,
        float* __restrict__ colsum, float* __restrict__ colsq,
        int doStats, int doRelu)
{
    __shared__ _Float16 lA[2][TILE_M * LDA_H];
    __shared__ _Float16 lB[2][32 * LDB_H];

    const int t      = threadIdx.x;
    const int lane   = t & 31;
    const int wave   = t >> 5;
    const int blockM = blockIdx.x * TILE_M;
    const int n0     = blockIdx.y * TILE_N;

    v8f acc[4] = {};

    // prologue: stage tile 0
    stage_tiles(X, ldx, W, NC, n0, lA[0], lB[0], t, blockM, M, 0);

    int cur = 0;
    for (int k0 = 0; k0 < K; k0 += 32) {
        const int nxt = k0 + 32;
        if (nxt < K) {
            // stage next tile into the other buffer, then wait until only
            // those 3 in-flight ops remain -> current tile is complete
            stage_tiles(X, ldx, W, NC, n0, lA[cur ^ 1], lB[cur ^ 1],
                        t, blockM, M, nxt);
            asm volatile("s_wait_asynccnt 0x3" ::: "memory");
        } else {
            asm volatile("s_wait_asynccnt 0x0" ::: "memory");
        }
        __syncthreads();   // RAW: all waves' current tile resident in LDS

        // ---- A fragment: lane<16 -> row=lane, K {0..7,16..23};
        //      lane>=16 -> row=lane-16, K {8..15,24..31} ----
        const int mrow = wave * 16 + (lane & 15);
        const _Float16* ap = &lA[cur][mrow * LDA_H + ((lane < 16) ? 0 : 8)];
        v16h a = cat8(*(const v8h*)ap, *(const v8h*)(ap + 16));

        // ---- all 4 B fragments up front (lane L holds B col N=L%16, K same order as A) ----
        v16h b[4];
        #pragma unroll
        for (int nt = 0; nt < 4; ++nt) {
            #pragma unroll
            for (int s = 0; s < 16; ++s) {
                const int kk = (s < 8) ? ((lane >> 4) * 8 + s) : (16 + (lane >> 4) * 8 + (s - 8));
                b[nt][s] = lB[cur][kk * LDB_H + nt * 16 + (lane & 15)];
            }
        }
        // ---- 4 back-to-back WMMAs sharing the A fragment ----
        #pragma unroll
        for (int nt = 0; nt < 4; ++nt)
            acc[nt] = __builtin_amdgcn_wmma_f32_16x16x32_f16(
                          false, a, false, b[nt], (short)0, acc[nt], false, false);

        __syncthreads();   // WAR: buffer 'cur' may be overwritten next iter
        cur ^= 1;
    }

    // ---- epilogue: C/D layout -> lane = N col, vgpr r = M row (+8 hi) ----
    #pragma unroll
    for (int nt = 0; nt < 4; ++nt) {
        const int col = n0 + nt * 16 + (lane & 15);
        const float bv = bias ? bias[col] : 0.0f;
        float s = 0.f, s2 = 0.f;
        #pragma unroll
        for (int r = 0; r < 8; ++r) {
            int row = blockM + wave * 16 + ((lane < 16) ? r : (8 + r));
            if (row < M) {
                float v = acc[nt][r] + bv;
                if (doRelu) v = fmaxf(v, 0.f);
                if (Y32) Y32[(size_t)row * ldy + yoff + col] = v;
                if (Y16) Y16[(size_t)row * ldy + yoff + col] = (_Float16)v;
                s  += v;
                s2 += v * v;
            }
        }
        if (doStats) {
            atomAddF32(&colsum[col], s);
            atomAddF32(&colsq[col],  s2);
        }
    }
}

// --------------------- BatchNorm (batch stats) + ReLU ----------------------
__global__ void bn_relu_k(const float* __restrict__ Y, int ldy,
                          const float* __restrict__ g, const float* __restrict__ be,
                          const float* __restrict__ cs, const float* __restrict__ cq,
                          _Float16* __restrict__ O, int ldo, int ooff,
                          int M, int d) {
    int idx = blockIdx.x * blockDim.x + threadIdx.x;
    if (idx >= M * d) return;
    int r = idx / d, c = idx % d;
    float invN = 1.0f / (float)M;
    float m = cs[c] * invN;
    float v = cq[c] * invN - m * m;
    float val = Y[(size_t)r * ldy + c];
    val = g[c] * (val - m) * rsqrtf(v + 1e-5f) + be[c];
    O[(size_t)r * ldo + ooff + c] = (_Float16)(val > 0.f ? val : 0.f);
}

// ----------------------- final 384 -> 1 GEMV + ReLU ------------------------
__global__ void gemv_relu_k(const _Float16* __restrict__ X, int ldx,
                            const float* __restrict__ w, const float* __restrict__ b,
                            float* __restrict__ out, int M, int K) {
    int r = blockIdx.x * blockDim.x + threadIdx.x;
    if (r >= M) return;
    float s = b[0];
    const _Float16* xp = X + (size_t)r * ldx;
    #pragma unroll 4
    for (int k = 0; k < K; ++k) s += (float)xp[k] * w[k];
    out[r] = s > 0.f ? s : 0.f;
}

// ---------------------------------------------------------------------------
extern "C" void kernel_launch(void* const* d_in, const int* in_sizes, int n_in,
                              void* d_out, int out_size, void* d_ws, size_t ws_size,
                              hipStream_t stream) {
    (void)n_in; (void)out_size; (void)ws_size;
    const int F = 64, H = 128, H3 = 384;
    const int N = in_sizes[0] / F;
    const int E = in_sizes[1] / 2;

    const float* x   = (const float*)d_in[0];
    const int*   src = (const int*)d_in[1];
    const int*   dst = src + E;

    // ---- workspace carve-out (64B-aligned regions) ----
    char* p = (char*)d_ws;
    auto carve = [&](size_t bytes) -> void* {
        void* r = (void*)p;
        p += (bytes + 63) & ~(size_t)63;
        return r;
    };
    _Float16* xc16  = (_Float16*)carve((size_t)N * H3 * 2); // x1|x2|x3 concat
    _Float16* a16   = (_Float16*)carve((size_t)N * H  * 2); // f16 GEMM A (conv)
    _Float16* b16   = (_Float16*)carve((size_t)N * H  * 2); // block-1 output
    _Float16* m16   = (_Float16*)carve((size_t)N * H3 * 2); // l1 output
    float*    agg32 = (float*)   carve((size_t)N * H  * 4); // h + neighbor sum
    float*    tmp32 = (float*)   carve((size_t)N * H3 * 4); // pre-BN GEMM out
    float*    cs    = (float*)   carve(H3 * 4);
    float*    cq    = (float*)   carve(H3 * 4);
    _Float16* w16   = (_Float16*)carve((size_t)H3 * H3 * 2); // f16 weights

    const int thr = 256;
    dim3 gridH((N + TILE_M - 1) / TILE_M, H  / TILE_N);
    dim3 grid3((N + TILE_M - 1) / TILE_M, H3 / TILE_N);

    for (int c = 0; c < 3; ++c) {
        const int din = (c == 0) ? F : H;
        const int wi  = 2 + c * 8;
        const float* w1  = (const float*)d_in[wi + 0];
        const float* b1  = (const float*)d_in[wi + 1];
        const float* g1  = (const float*)d_in[wi + 2];
        const float* be1 = (const float*)d_in[wi + 3];
        const float* w2  = (const float*)d_in[wi + 4];
        const float* b2  = (const float*)d_in[wi + 5];
        const float* g2  = (const float*)d_in[wi + 6];
        const float* be2 = (const float*)d_in[wi + 7];

        // h2 = h ; h2[dst] += h[src]   (fp32 accumulate)
        if (c == 0) {
            copy_k<float><<<(N * din + thr - 1) / thr, thr, 0, stream>>>(
                x, F, 0, agg32, din, N, din);
            scatter_k<float><<<(E * (din / 4) + thr - 1) / thr, thr, 0, stream>>>(
                x, F, 0, src, dst, agg32, din, E, din / 4);
        } else {
            copy_k<_Float16><<<(N * din + thr - 1) / thr, thr, 0, stream>>>(
                xc16, H3, (c - 1) * H, agg32, din, N, din);
            scatter_k<_Float16><<<(E * (din / 4) + thr - 1) / thr, thr, 0, stream>>>(
                xc16, H3, (c - 1) * H, src, dst, agg32, din, E, din / 4);
        }
        // f16 operand for WMMA
        cvt_k<<<(N * din + thr - 1) / thr, thr, 0, stream>>>(agg32, a16, N * din);

        // block 1: relu(bn(h2 @ w1 + b1))
        cvt_k<<<(din * H + thr - 1) / thr, thr, 0, stream>>>(w1, w16, din * H);
        zero_k<<<(2 * H3 + thr - 1) / thr, thr, 0, stream>>>(cs, 2 * H3);
        wmma_gemm16<<<gridH, thr, 0, stream>>>(
            a16, din, w16, b1, tmp32, nullptr, H, 0, N, din, H, cs, cq, 1, 0);
        bn_relu_k<<<(N * H + thr - 1) / thr, thr, 0, stream>>>(
            tmp32, H, g1, be1, cs, cq, b16, H, 0, N, H);

        // block 2: relu(bn(. @ w2 + b2)) -> xc16 column slice c
        cvt_k<<<(H * H + thr - 1) / thr, thr, 0, stream>>>(w2, w16, H * H);
        zero_k<<<(2 * H3 + thr - 1) / thr, thr, 0, stream>>>(cs, 2 * H3);
        wmma_gemm16<<<gridH, thr, 0, stream>>>(
            b16, H, w16, b2, tmp32, nullptr, H, 0, N, H, H, cs, cq, 1, 0);
        bn_relu_k<<<(N * H + thr - 1) / thr, thr, 0, stream>>>(
            tmp32, H, g2, be2, cs, cq, xc16, H3, c * H, N, H);
    }

    const float* l1_w = (const float*)d_in[26];
    const float* l1_b = (const float*)d_in[27];
    const float* l2_w = (const float*)d_in[28];
    const float* l2_b = (const float*)d_in[29];
    const float* o_w  = (const float*)d_in[30];
    const float* o_b  = (const float*)d_in[31];

    // h = relu(xc @ l1_w + l1_b)  -> m16 (f16)
    cvt_k<<<(H3 * H3 + thr - 1) / thr, thr, 0, stream>>>(l1_w, w16, H3 * H3);
    wmma_gemm16<<<grid3, thr, 0, stream>>>(
        xc16, H3, w16, l1_b, nullptr, m16, H3, 0, N, H3, H3,
        nullptr, nullptr, 0, 1);
    // h = relu(h @ l2_w + l2_b)   -> xc16 (reuse)
    cvt_k<<<(H3 * H3 + thr - 1) / thr, thr, 0, stream>>>(l2_w, w16, H3 * H3);
    wmma_gemm16<<<grid3, thr, 0, stream>>>(
        m16, H3, w16, l2_b, nullptr, xc16, H3, 0, N, H3, H3,
        nullptr, nullptr, 0, 1);
    // out = relu(h @ o_w + o_b)
    gemv_relu_k<<<(N + thr - 1) / thr, thr, 0, stream>>>(
        xc16, H3, o_w, o_b, (float*)d_out, N, H3);
}
